// LlamaCosAttention_7232724927157
// MI455X (gfx1250) — matrix operand order repaired and checked by reading, not compiled, not run on hardware
//
#include <hip/hip_runtime.h>
#include <hip/hip_bf16.h>

typedef __bf16 bf16_t;
typedef __attribute__((ext_vector_type(16))) bf16_t v16bf;
typedef __attribute__((ext_vector_type(8))) float v8f;
typedef __attribute__((ext_vector_type(4))) unsigned int v4u;
typedef __attribute__((ext_vector_type(8))) int v8i_t;
typedef __attribute__((ext_vector_type(4))) int v4i_t;

#define WMMA_BF16(A_, B_, C_) \
  __builtin_amdgcn_wmma_f32_16x16x32_bf16(false, (A_), false, (B_), (short)0, (C_), false, false)

__device__ __forceinline__ bf16_t f2bf(float f) {
  unsigned u = __builtin_bit_cast(unsigned, f);
  u += 0x7FFFu + ((u >> 16) & 1u);
  unsigned short s = (unsigned short)(u >> 16);
  return __builtin_bit_cast(bf16_t, s);
}
__device__ __forceinline__ float bf2f(bf16_t h) {
  unsigned short s = __builtin_bit_cast(unsigned short, h);
  unsigned u = ((unsigned)s) << 16;
  return __builtin_bit_cast(float, u);
}
__device__ __forceinline__ float halfmax16(float x) {
#pragma unroll
  for (int m = 1; m < 16; m <<= 1) x = fmaxf(x, __shfl_xor(x, m, 32));
  return x;
}
__device__ __forceinline__ float halfsum16(float x) {
#pragma unroll
  for (int m = 1; m < 16; m <<= 1) x += __shfl_xor(x, m, 32);
  return x;
}

// ---- Tensor Data Mover: 2D bf16 tile (tile_w x tile_h elems) global -> LDS ----
// LDS rows are written with TDM padding so they land at stride (tile_w + 8) bf16.
// pad_icode: row bytes = 2^(pad_icode+1) DWORDs; pad_acode: pad = (pad_acode+1) DWORDs.
__device__ __forceinline__ void tdm_load_2d(unsigned lds_off, const bf16_t* gptr,
                                            unsigned tensor_w, unsigned tensor_h,
                                            unsigned tile_w, unsigned tile_h,
                                            unsigned stride_elems,
                                            unsigned pad_icode, unsigned pad_acode) {
  unsigned long long ga = (unsigned long long)(uintptr_t)gptr;
  v4u g0 = {0x1u,  // count=1, user descriptor
            lds_off,
            (unsigned)(ga & 0xffffffffu),
            (unsigned)((ga >> 32) & 0x01ffffffu) | (2u << 30)};  // type=2 (image)
  unsigned d0 = (1u << 16)            // data_size = 2 bytes
              | (1u << 20)            // pad_enable
              | (pad_icode << 22)     // pad_interval
              | (pad_acode << 25);    // pad_amount
  v8i_t g1 = {(int)d0,
              (int)((tensor_w & 0xffffu) << 16),
              (int)(((tensor_w >> 16) & 0xffffu) | ((tensor_h & 0xffffu) << 16)),
              (int)(((tensor_h >> 16) & 0xffffu) | ((tile_w & 0xffffu) << 16)),
              (int)(tile_h & 0xffffu),  // tile_dim1; tile_dim2 = 0
              (int)stride_elems,        // tensor_dim0_stride[31:0]
              0, 0};
  v4i_t z4 = {0, 0, 0, 0};
#if defined(__clang_major__) && __clang_major__ >= 23
  v8i_t z8 = {0, 0, 0, 0, 0, 0, 0, 0};
  __builtin_amdgcn_tensor_load_to_lds(g0, g1, z4, z4, z8, 0);
#else
  __builtin_amdgcn_tensor_load_to_lds(g0, g1, z4, z4, 0);
#endif
}

// ---------------- elementwise f32 -> bf16 ----------------
__global__ void cvt_f32_bf16_kernel(const float* __restrict__ in, bf16_t* __restrict__ out, int n) {
  int idx = blockIdx.x * blockDim.x + threadIdx.x;
  if (idx < n) out[idx] = f2bf(in[idx]);
}

// ---------------- per-token per-head gates: sigmoid(h @ Wg + bg) ----------------
__global__ void gates_kernel(const float* __restrict__ hid,
                             const float* __restrict__ Wg1, const float* __restrict__ bg1,
                             const float* __restrict__ Wg2, const float* __restrict__ bg2,
                             float* __restrict__ g1, float* __restrict__ g2, int M) {
  int idx = blockIdx.x * blockDim.x + threadIdx.x;
  if (idx >= M * 32) return;
  int m = idx >> 5;
  int j = idx & 31;
  int h = j & 15;
  bool second = (j >= 16);
  const float* W = second ? Wg2 : Wg1;
  const float* bias = second ? bg2 : bg1;
  const float* row = hid + (size_t)m * 2048;
  float acc = 0.f;
  for (int k = 0; k < 2048; ++k) acc += row[k] * W[(size_t)k * 16 + h];
  acc += bias[h];
  float sg = 1.f / (1.f + __expf(-acc));
  (second ? g2 : g1)[(size_t)m * 16 + h] = sg;
}

// ---------------- RoPE (f32 in -> bf16 out) ----------------
__global__ void rope_kernel(const float* __restrict__ in, const int* __restrict__ pos_ids,
                            bf16_t* __restrict__ out, int M, int W, int S) {
  int idx = blockIdx.x * blockDim.x + threadIdx.x;
  if (idx >= M * W) return;
  int m = idx / W;
  int c = idx - m * W;
  int d = c & 127;
  int s = m % S;
  float pos = (float)pos_ids[s];
  int j = d & 63;
  float inv = __powf(10000.f, -(float)j * (1.f / 64.f));
  float ang = pos * inv;
  float cv = __cosf(ang), sv = __sinf(ang);
  float x = in[idx];
  float rot = (d < 64) ? -in[idx + 64] : in[idx - 64];
  out[idx] = f2bf(x * cv + rot * sv);
}

// ---------------- tiled bf16 GEMM with TDM double-buffered tiles ----------------
// C_f32[M,N] = A_bf16[M,K] * B_bf16[K,N]
__global__ __launch_bounds__(256) void gemm_bf16_kernel(const bf16_t* __restrict__ A,
                                                        const bf16_t* __restrict__ Bw,
                                                        float* __restrict__ C,
                                                        int M, int N, int K) {
  constexpr int AS = 40;  // 64x32 A tile rows: 64B data + 16B TDM pad
  constexpr int BS = 72;  // 32x64 B tile rows: 128B data + 16B TDM pad
  __shared__ __align__(16) bf16_t As[2 * 64 * AS];
  __shared__ __align__(16) bf16_t Bs[2 * 32 * BS];

  int t = threadIdx.x;
  int lane = t & 31, wave = t >> 5;
  int hf = lane >> 4, r = lane & 15;
  int wm = wave >> 2, wn = wave & 3;  // wave grid 2 (M) x 4 (N)
  int m0 = blockIdx.y * 64, n0 = blockIdx.x * 64;

  unsigned asOff = (unsigned)(uintptr_t)&As[0];
  unsigned bsOff = (unsigned)(uintptr_t)&Bs[0];
  const unsigned asBuf = 64 * AS * 2;  // bytes per A buffer
  const unsigned bsBuf = 32 * BS * 2;  // bytes per B buffer

  v8f acc0 = {}, acc1 = {};
  int Nk = K >> 5;

  if (wave == 0) {
    // prologue: stage tile 0 into buffer 0
    tdm_load_2d(asOff, A + (size_t)m0 * K, (unsigned)K, (unsigned)M, 32, 64, (unsigned)K, 3, 3);
    tdm_load_2d(bsOff, Bw + n0, (unsigned)N, (unsigned)K, 64, 32, (unsigned)N, 4, 3);
  }

  for (int kt = 0; kt < Nk; ++kt) {
    int buf = kt & 1;
    if (wave == 0) {
      if (kt + 1 < Nk) {
        int k1 = (kt + 1) << 5;
        tdm_load_2d(asOff + (buf ^ 1) * asBuf, A + (size_t)m0 * K + k1,
                    (unsigned)K, (unsigned)M, 32, 64, (unsigned)K, 3, 3);
        tdm_load_2d(bsOff + (buf ^ 1) * bsBuf, Bw + (size_t)k1 * N + n0,
                    (unsigned)N, (unsigned)K, 64, 32, (unsigned)N, 4, 3);
        __builtin_amdgcn_s_wait_tensorcnt((short)2);  // pair kt complete (in-order)
      } else {
        __builtin_amdgcn_s_wait_tensorcnt((short)0);
      }
    }
    __syncthreads();  // tile kt visible to all waves

    const bf16_t* Ab = &As[buf * 64 * AS];
    const bf16_t* Bb = &Bs[buf * 32 * BS];
    v16bf af0, af1, bf;
#pragma unroll
    for (int e = 0; e < 16; ++e) {
      int ka = (e & 7) + 8 * hf + 16 * (e >> 3);  // A-operand K pattern
      int kb = (e & 7) + 8 * hf + 16 * (e >> 3);  // B-operand K pattern
      af0[e] = Ab[(wm * 32 + r) * AS + ka];
      af1[e] = Ab[(wm * 32 + 16 + r) * AS + ka];
      bf[e] = Bb[kb * BS + wn * 16 + r];
    }
    acc0 = WMMA_BF16(af0, bf, acc0);
    acc1 = WMMA_BF16(af1, bf, acc1);

    __syncthreads();  // all reads of buf done before DMA reuses it
  }

#pragma unroll
  for (int v = 0; v < 8; ++v) {
    int row0 = m0 + wm * 32 + v + 8 * hf;
    int col = n0 + wn * 16 + r;
    C[(size_t)row0 * N + col] = acc0[v];
    C[(size_t)(row0 + 16) * N + col] = acc1[v];
  }
}

// ---------------- flash attention + value gate + per-head LN + output gate ----------------
// grid: (S/128, B*16heads); block 256 (8 waves); wave owns 16 q rows.
__global__ __launch_bounds__(256) void attn_kernel(const bf16_t* __restrict__ Q,
                                                   const bf16_t* __restrict__ Kb,
                                                   const bf16_t* __restrict__ Vb,
                                                   const float* __restrict__ g1,
                                                   const float* __restrict__ g2,
                                                   const float* __restrict__ gamma,
                                                   const float* __restrict__ beta,
                                                   bf16_t* __restrict__ Out, int S) {
  constexpr int QS = 136, KS = 136, VS = 136, PS = 40;
  __shared__ __align__(16) bf16_t Qs[8 * 16 * QS];
  __shared__ __align__(16) bf16_t Ks[32 * KS];
  __shared__ __align__(16) bf16_t Vs[32 * VS];
  __shared__ __align__(16) bf16_t Pss[8 * 16 * PS];

  int t = threadIdx.x, lane = t & 31, wave = t >> 5;
  int hf = lane >> 4, r = lane & 15;
  int batch = blockIdx.y >> 4, head = blockIdx.y & 15, kvh = head >> 2;
  int qblock = blockIdx.x * 128;
  int qbase = qblock + wave * 16;
  size_t base = (size_t)batch * S;

  bf16_t* Qw = &Qs[wave * 16 * QS];
  bf16_t* Pw = &Pss[wave * 16 * PS];
  unsigned ksOff = (unsigned)(uintptr_t)&Ks[0];

  // per-wave Q tile load (16 q rows x 128 d), intra-wave DS ordering -> no barrier needed
#pragma unroll
  for (int j = 0; j < 8; ++j) {
    int lin = lane * 64 + j * 8;
    int row = lin >> 7, col = lin & 127;
    *(int4*)&Qw[row * QS + col] =
        *(const int4*)&Q[(base + qbase + row) * 2048 + head * 128 + col];
  }

  // Q A-fragments (4 chunks of K=32 over head_dim=128)
  v16bf qf[4];
#pragma unroll
  for (int c = 0; c < 4; ++c)
#pragma unroll
    for (int e = 0; e < 16; ++e) {
      int ka = (e & 7) + 8 * hf + 16 * (e >> 3);
      qf[c][e] = Qw[r * QS + c * 32 + ka];
    }

  v8f o[8] = {};
  float mrun[8], lrun[8];
#pragma unroll
  for (int v = 0; v < 8; ++v) { mrun[v] = -3e38f; lrun[v] = 0.f; }

  const float scale = 0.08838834764831845f;  // 1/sqrt(128)
  int nkb = (qblock + 128) >> 5;

  for (int kb = 0; kb < nkb; ++kb) {
    int kbase = kb * 32;
    __syncthreads();  // previous iteration's reads of Ks/Vs finished

    // K tile (32 keys x 128 d) via Tensor Data Mover, issued by wave 0
    if (wave == 0) {
      tdm_load_2d(ksOff, Kb + (base + kbase) * 512 + kvh * 128,
                  512u, (unsigned)(2 * S), 128, 32, 512u, 5, 3);
    }
    // V tile: manual copy so the per-(key,head) gate g2 can be folded in
#pragma unroll
    for (int j = 0; j < 2; ++j) {
      int lin = t * 16 + j * 8;
      int row = lin >> 7, col = lin & 127;
      int4 vd = *(const int4*)&Vb[(base + kbase + row) * 512 + kvh * 128 + col];
      alignas(16) bf16_t tmp[8];
      alignas(16) bf16_t gv8[8];
      *(int4*)tmp = vd;
      float gv = g2[(base + kbase + row) * 16 + head];
#pragma unroll
      for (int u = 0; u < 8; ++u) gv8[u] = f2bf(bf2f(tmp[u]) * gv);
      *(int4*)&Vs[row * VS + col] = *(int4*)gv8;
    }
    if (wave == 0) __builtin_amdgcn_s_wait_tensorcnt((short)0);
    __syncthreads();

    if (kbase <= qbase + 15) {
      // S = Q * K^T  (two 16x16 key tiles)
      v8f s0 = {}, s1 = {};
#pragma unroll
      for (int c = 0; c < 4; ++c) {
        v16bf kf0, kf1;
#pragma unroll
        for (int e = 0; e < 16; ++e) {
          int kbp = (e & 7) + 8 * hf + 16 * (e >> 3);  // B-operand contraction index (d within chunk)
          kf0[e] = Ks[r * KS + c * 32 + kbp];
          kf1[e] = Ks[(16 + r) * KS + c * 32 + kbp];
        }
        s0 = WMMA_BF16(qf[c], kf0, s0);
        s1 = WMMA_BF16(qf[c], kf1, s1);
      }

      // online softmax, causal mask; write P through per-wave LDS scratch
#pragma unroll
      for (int v = 0; v < 8; ++v) {
        int qg = qbase + v + 8 * hf;
        int c0 = kbase + r, c1 = kbase + 16 + r;
        float x0 = s0[v] * scale, x1 = s1[v] * scale;
        bool msk0 = (c0 > qg), msk1 = (c1 > qg);
        float rm = halfmax16(fmaxf(msk0 ? -3e38f : x0, msk1 ? -3e38f : x1));
        float mn = fmaxf(mrun[v], rm);
        float p0 = msk0 ? 0.f : __expf(x0 - mn);
        float p1 = msk1 ? 0.f : __expf(x1 - mn);
        float al = __expf(mrun[v] - mn);
        mrun[v] = mn;
        lrun[v] = lrun[v] * al + halfsum16(p0 + p1);
#pragma unroll
        for (int f = 0; f < 8; ++f) o[f][v] *= al;
        Pw[(v + 8 * hf) * PS + r] = f2bf(p0);
        Pw[(v + 8 * hf) * PS + 16 + r] = f2bf(p1);
      }

      // O += P * V
      v16bf pf;
#pragma unroll
      for (int e = 0; e < 16; ++e) {
        int ka = (e & 7) + 8 * hf + 16 * (e >> 3);
        pf[e] = Pw[r * PS + ka];
      }
#pragma unroll
      for (int f = 0; f < 8; ++f) {
        v16bf vf;
#pragma unroll
        for (int e = 0; e < 16; ++e) {
          int kbp = (e & 7) + 8 * hf + 16 * (e >> 3);
          vf[e] = Vs[kbp * VS + f * 16 + r];
        }
        o[f] = WMMA_BF16(pf, vf, o[f]);
      }
    }
  }

  // epilogue: 1/l, per-head LayerNorm over d=128, gamma/beta, output gate g1
  float gam[8], bet[8];
#pragma unroll
  for (int f = 0; f < 8; ++f) { gam[f] = gamma[f * 16 + r]; bet[f] = beta[f * 16 + r]; }

#pragma unroll
  for (int v = 0; v < 8; ++v) {
    float iv = 1.0f / lrun[v];
    float s = 0.f, s2 = 0.f;
#pragma unroll
    for (int f = 0; f < 8; ++f) {
      float x = o[f][v] * iv;
      o[f][v] = x;
      s += x;
      s2 += x * x;
    }
    s = halfsum16(s);
    s2 = halfsum16(s2);
    float mean = s * (1.f / 128.f);
    float var = s2 * (1.f / 128.f) - mean * mean;
    float rstd = rsqrtf(var + 1e-5f);
    int qg = qbase + v + 8 * hf;
    float g1v = g1[(base + qg) * 16 + head];
#pragma unroll
    for (int f = 0; f < 8; ++f) {
      float y = ((o[f][v] - mean) * rstd * gam[f] + bet[f]) * g1v;
      Out[(base + qg) * 2048 + head * 128 + f * 16 + r] = f2bf(y);
    }
  }
}

// ---------------- host-side orchestration ----------------
extern "C" void kernel_launch(void* const* d_in, const int* in_sizes, int n_in,
                              void* d_out, int out_size, void* d_ws, size_t ws_size,
                              hipStream_t stream) {
  (void)in_sizes; (void)n_in; (void)out_size; (void)ws_size;
  const float* hidden = (const float*)d_in[0];
  const int* pos = (const int*)d_in[1];
  const float* Wq = (const float*)d_in[2];
  const float* Wk = (const float*)d_in[3];
  const float* Wv = (const float*)d_in[4];
  const float* Wo = (const float*)d_in[5];
  const float* Wg1 = (const float*)d_in[6];
  const float* bg1 = (const float*)d_in[7];
  const float* Wg2 = (const float*)d_in[8];
  const float* bg2 = (const float*)d_in[9];
  const float* gamma = (const float*)d_in[10];
  const float* beta = (const float*)d_in[11];
  float* out = (float*)d_out;

  const int Bsz = 2, S = 2048, H = 2048;
  const int M = Bsz * S;  // 4096 tokens
  const int QD = 2048, KVD = 512;

  char* w = (char*)d_ws;
  auto alloc = [&](size_t bytes) {
    char* p = w;
    w += (bytes + 255) & ~(size_t)255;
    return p;
  };
  bf16_t* hid_bf = (bf16_t*)alloc((size_t)M * H * 2);
  bf16_t* Wq_bf = (bf16_t*)alloc((size_t)H * QD * 2);
  bf16_t* Wk_bf = (bf16_t*)alloc((size_t)H * KVD * 2);
  bf16_t* Wv_bf = (bf16_t*)alloc((size_t)H * KVD * 2);
  bf16_t* Wo_bf = (bf16_t*)alloc((size_t)QD * H * 2);
  float* qf32 = (float*)alloc((size_t)M * QD * 4);
  float* kf32 = (float*)alloc((size_t)M * KVD * 4);
  float* vf32 = (float*)alloc((size_t)M * KVD * 4);
  bf16_t* q_bf = (bf16_t*)alloc((size_t)M * QD * 2);
  bf16_t* k_bf = (bf16_t*)alloc((size_t)M * KVD * 2);
  bf16_t* v_bf = (bf16_t*)alloc((size_t)M * KVD * 2);
  float* g1 = (float*)alloc((size_t)M * 16 * 4);
  float* g2 = (float*)alloc((size_t)M * 16 * 4);
  bf16_t* attn_bf = (bf16_t*)alloc((size_t)M * QD * 2);

  auto cvt = [&](const float* src, bf16_t* dst, int n) {
    cvt_f32_bf16_kernel<<<(n + 255) / 256, 256, 0, stream>>>(src, dst, n);
  };

  // 1) downconvert activations + weights to bf16
  cvt(hidden, hid_bf, M * H);
  cvt(Wq, Wq_bf, H * QD);
  cvt(Wk, Wk_bf, H * KVD);
  cvt(Wv, Wv_bf, H * KVD);
  cvt(Wo, Wo_bf, QD * H);

  // 2) gates (small GEMV, f32)
  gates_kernel<<<(M * 32 + 255) / 256, 256, 0, stream>>>(hidden, Wg1, bg1, Wg2, bg2, g1, g2, M);

  // 3) QKV projections via WMMA (TDM-fed tiles)
  gemm_bf16_kernel<<<dim3(QD / 64, M / 64), 256, 0, stream>>>(hid_bf, Wq_bf, qf32, M, QD, H);
  gemm_bf16_kernel<<<dim3(KVD / 64, M / 64), 256, 0, stream>>>(hid_bf, Wk_bf, kf32, M, KVD, H);
  gemm_bf16_kernel<<<dim3(KVD / 64, M / 64), 256, 0, stream>>>(hid_bf, Wv_bf, vf32, M, KVD, H);

  // 4) RoPE on q/k (-> bf16); plain convert for v
  rope_kernel<<<(M * QD + 255) / 256, 256, 0, stream>>>(qf32, pos, q_bf, M, QD, S);
  rope_kernel<<<(M * KVD + 255) / 256, 256, 0, stream>>>(kf32, pos, k_bf, M, KVD, S);
  cvt(vf32, v_bf, M * KVD);

  // 5) flash attention + value gate + per-head LN + output gate
  attn_kernel<<<dim3(S / 128, Bsz * 16), 256, 0, stream>>>(q_bf, k_bf, v_bf, g1, g2, gamma,
                                                           beta, attn_bf, S);

  // 6) output projection via WMMA (f32 out)
  gemm_bf16_kernel<<<dim3(H / 64, M / 64), 256, 0, stream>>>(attn_bf, Wo_bf, out, M, H, QD);
}
